// GNMT_11089605558604
// MI455X (gfx1250) — matrix operand order repaired and checked by reading, not compiled, not run on hardware
//
#include <hip/hip_runtime.h>
#include <cstdint>
#include <cstddef>

// ---------------------------------------------------------------------------
// Types for CDNA5 WMMA (gfx1250, wave32)
// ---------------------------------------------------------------------------
typedef unsigned short u16;
typedef __attribute__((ext_vector_type(16))) __bf16        v16bf;
typedef __attribute__((ext_vector_type(8)))  float         v8f;
typedef __attribute__((ext_vector_type(4)))  unsigned int  u32x4;

union BF16x16 { v16bf v; u32x4 q[2]; u16 s[16]; };

__device__ inline u16 f2bf(float f) {
    unsigned u = __builtin_bit_cast(unsigned, f);
    unsigned r = (u + 0x7FFFu + ((u >> 16) & 1u)) >> 16;   // round-to-nearest-even
    return (u16)r;
}
__device__ inline float bf2f(u16 h) {
    return __builtin_bit_cast(float, ((unsigned)h) << 16);
}
__device__ inline float sigf(float x) { return 1.0f / (1.0f + __expf(-x)); }

__device__ inline v8f wmma_bf16(v16bf a, v16bf b, v8f c) {
    // v_wmma_f32_16x16x32_bf16  D = A(16x32) * B(32x16) + C
    return __builtin_amdgcn_wmma_f32_16x16x32_bf16(false, a, false, b, (short)0, c, false, false);
}

// A fragment (16x32, rows = M): ISA layout — lanes 0-15 row M=lane hold K {kb..kb+7, kb+16..kb+23},
// lanes 16-31 row M=lane-16 hold K {kb+8..kb+15, kb+24..kb+31}.
__device__ inline v16bf load_frag_a(const u16* base, int lda, int m0, int kb, int lane) {
    int hl = lane >> 4, mr = lane & 15;
    const u16* p = base + (size_t)(m0 + mr) * lda + kb + hl * 8;
    BF16x16 t;
    t.q[0] = *(const u32x4*)(p);
    t.q[1] = *(const u32x4*)(p + 16);
    return t.v;
}
// B fragment (32x16, cols = N) from row-major weight W[N][K] (computing X @ W^T):
// lanes 0-15: col N=n0+lane, K {kb..kb+7, kb+16..kb+23} ; lanes 16-31: col N=n0+lane-16, K {kb+8..kb+15, kb+24..kb+31}.
// two 16-byte chunks per lane.
__device__ inline v16bf load_frag_b(const u16* W, int ldb, int n0, int kb, int lane) {
    int hl = lane >> 4, nr = lane & 15;
    const u16* p = W + (size_t)(n0 + nr) * ldb + kb + hl * 8;
    BF16x16 t;
    t.q[0] = *(const u32x4*)(p);
    t.q[1] = *(const u32x4*)(p + 16);
    return t.v;
}

// ---------------------------------------------------------------------------
// Generic big GEMM: C(MxN,f32) = A(MxK,bf16,row-major) @ W(NxK,bf16,row-major)^T + bias
// One 16x16 output tile per wave. remap!=0: output row r=t*16+b -> row b*64+t (B=16,T=64).
// ---------------------------------------------------------------------------
__global__ void gemm_bf16(const u16* __restrict__ A, int lda,
                          const u16* __restrict__ W, int ldb,
                          const float* __restrict__ bias,
                          float* __restrict__ C, int ldc,
                          int M, int N, int K, int remap)
{
    int lane = threadIdx.x & 31;
    int g = blockIdx.x * (blockDim.x >> 5) + (threadIdx.x >> 5);
    int tn = N >> 4;
    int total = (M >> 4) * tn;
    if (g >= total) return;
    int m0 = (g / tn) << 4, n0 = (g % tn) << 4;
    int hl = lane >> 4, nr = lane & 15;

    v8f acc;
    float bv = bias ? bias[n0 + nr] : 0.0f;
    for (int j = 0; j < 8; ++j) acc[j] = bv;

    for (int kb = 0; kb < K; kb += 32) {
        v16bf a = load_frag_a(A, lda, m0, kb, lane);
        v16bf b = load_frag_b(W, ldb, n0, kb, lane);
        acc = wmma_bf16(a, b, acc);
    }
    for (int j = 0; j < 8; ++j) {
        int r = m0 + j + 8 * hl;                      // D layout: vgpr j -> row j / j+8
        int c = n0 + nr;
        int orow = remap ? ((r & 15) * 64 + (r >> 4)) : r;
        C[(size_t)orow * ldc + c] = acc[j];
    }
}

// ---------------------------------------------------------------------------
// Encoder LSTM scan: persistent single workgroup (32 waves), h in LDS, c in regs.
// gates(t) = gx[t] (precomputed X@Wih^T + bias, f32) + h(t-1) @ Whh^T  (WMMA)
// Wave w owns gate columns [16w,16w+16) of all four gates i/f/g/o.
// ---------------------------------------------------------------------------
__global__ void __launch_bounds__(1024)
enc_scan(const float* __restrict__ gx, const u16* __restrict__ whh,
         u16* __restrict__ ys, int ldy, int col0, int T, int rev)
{
    __shared__ u16 hbuf[16 * 512];
    int tid = threadIdx.x, lane = tid & 31, w = tid >> 5;
    int hl = lane >> 4, nr = lane & 15;

    for (int i = tid; i < 16 * 512; i += 1024) hbuf[i] = 0;
    v8f cfrag;
    for (int j = 0; j < 8; ++j) cfrag[j] = 0.0f;
    __syncthreads();

    for (int s = 0; s < T; ++s) {
        int t = rev ? (T - 1 - s) : s;
        v8f acc[4];
        for (int ggt = 0; ggt < 4; ++ggt)
            for (int j = 0; j < 8; ++j)
                acc[ggt][j] = gx[(size_t)(t * 16 + j + 8 * hl) * 2048 + 512 * ggt + 16 * w + nr];

        for (int kb = 0; kb < 512; kb += 32) {
            v16bf a = load_frag_a(hbuf, 512, 0, kb, lane);
            for (int ggt = 0; ggt < 4; ++ggt) {
                v16bf b = load_frag_b(whh, 512, 512 * ggt + 16 * w, kb, lane);
                acc[ggt] = wmma_bf16(a, b, acc[ggt]);
            }
        }
        float hr[8];
        for (int j = 0; j < 8; ++j) {
            float iv = acc[0][j], fv = acc[1][j], gv = acc[2][j], ov = acc[3][j];
            float cv = sigf(fv) * cfrag[j] + sigf(iv) * tanhf(gv);
            cfrag[j] = cv;
            hr[j] = sigf(ov) * tanhf(cv);
        }
        __syncthreads();   // all waves finished reading h(t-1)
        for (int j = 0; j < 8; ++j) {
            int row = j + 8 * hl, col = 16 * w + nr;
            u16 hb = f2bf(hr[j]);
            hbuf[row * 512 + col] = hb;
            ys[(size_t)(t * 16 + row) * ldy + col0 + col] = hb;
        }
        __syncthreads();   // h(t) visible before next step
    }
}

// ---------------------------------------------------------------------------
// Decoder LSTM cell: fused-concat GEMM over K = nseg*512 with weight Wcat(2048 x K).
// Segments s0,s1,s2 are 16x512 bf16 (LDS). c in registers, h -> LDS (+ optional global).
// ---------------------------------------------------------------------------
__device__ inline void lstm_cell_dec(const u16* s0, const u16* s1, const u16* s2, int nseg,
                                     const u16* __restrict__ W, int ldb,
                                     const float* __restrict__ bias,
                                     v8f& cfrag, u16* hout, u16* gout, int lane, int w)
{
    int hl = lane >> 4, nr = lane & 15;
    v8f acc[4];
    for (int ggt = 0; ggt < 4; ++ggt) {
        float bv = bias[512 * ggt + 16 * w + nr];
        for (int j = 0; j < 8; ++j) acc[ggt][j] = bv;
    }
    int K = nseg * 512;
    for (int kb = 0; kb < K; kb += 32) {
        const u16* seg = (kb < 512) ? s0 : ((kb < 1024) ? s1 : s2);
        v16bf a = load_frag_a(seg, 512, 0, kb & 511, lane);
        for (int ggt = 0; ggt < 4; ++ggt) {
            v16bf b = load_frag_b(W, ldb, 512 * ggt + 16 * w, kb, lane);
            acc[ggt] = wmma_bf16(a, b, acc[ggt]);
        }
    }
    float hr[8];
    for (int j = 0; j < 8; ++j) {
        float iv = acc[0][j], fv = acc[1][j], gv = acc[2][j], ov = acc[3][j];
        float cv = sigf(fv) * cfrag[j] + sigf(iv) * tanhf(gv);
        cfrag[j] = cv;
        hr[j] = sigf(ov) * tanhf(cv);
    }
    __syncthreads();
    for (int j = 0; j < 8; ++j) {
        int row = j + 8 * hl, col = 16 * w + nr;
        u16 hb = f2bf(hr[j]);
        hout[row * 512 + col] = hb;
        if (gout) gout[row * 512 + col] = hb;
    }
    __syncthreads();
}

// ---------------------------------------------------------------------------
// Decoder: persistent single workgroup. 4 WMMA cells + WMMA q-proj + VALU attention.
// ---------------------------------------------------------------------------
__global__ void __launch_bounds__(1024)
decoder_kernel(const u16* __restrict__ embt,
               const u16* w0, const u16* w1, const u16* w2, const u16* w3,
               const float* b0, const float* b1, const float* b2, const float* b3,
               const u16* __restrict__ wqw, const float* __restrict__ kproj,
               const u16* __restrict__ encout, const int* __restrict__ src_len,
               u16* __restrict__ hs_top)
{
    __shared__ u16  embs[16 * 512];
    __shared__ u16  ctxs[16 * 512];
    __shared__ u16  hbuf[4][16 * 512];
    __shared__ u16  qbuf[16 * 512];
    __shared__ float sc[16 * 64];
    __shared__ float wts[16 * 64];

    int tid = threadIdx.x, lane = tid & 31, w = tid >> 5;
    int hl = lane >> 4, nr = lane & 15;

    for (int i = tid; i < 16 * 512; i += 1024) {
        ctxs[i] = 0;
        hbuf[0][i] = 0; hbuf[1][i] = 0; hbuf[2][i] = 0; hbuf[3][i] = 0;
    }
    v8f creg[4];
    for (int l = 0; l < 4; ++l)
        for (int j = 0; j < 8; ++j) creg[l][j] = 0.0f;
    __syncthreads();

    for (int t = 0; t < 64; ++t) {
        for (int i = tid; i < 8192; i += 1024) embs[i] = embt[(size_t)t * 8192 + i];
        __syncthreads();

        // layer0: A = [emb | ctx | h0], K=1536 ; layers 1-3: A = [h_{l-1} | h_l], K=1024
        lstm_cell_dec(embs, ctxs, hbuf[0], 3, w0, 1536, b0, creg[0], hbuf[0], nullptr, lane, w);
        lstm_cell_dec(hbuf[0], hbuf[1], nullptr, 2, w1, 1024, b1, creg[1], hbuf[1], nullptr, lane, w);
        lstm_cell_dec(hbuf[1], hbuf[2], nullptr, 2, w2, 1024, b2, creg[2], hbuf[2], nullptr, lane, w);
        lstm_cell_dec(hbuf[2], hbuf[3], nullptr, 2, w3, 1024, b3, creg[3], hbuf[3],
                      hs_top + (size_t)t * 8192, lane, w);

        // q = h3 @ Wq^T  (16x512, K=512) — one tile per wave
        v8f qa;
        for (int j = 0; j < 8; ++j) qa[j] = 0.0f;
        for (int kb = 0; kb < 512; kb += 32) {
            v16bf a = load_frag_a(hbuf[3], 512, 0, kb, lane);
            v16bf b = load_frag_b(wqw, 512, 16 * w, kb, lane);
            qa = wmma_bf16(a, b, qa);
        }
        for (int j = 0; j < 8; ++j)
            qbuf[(j + 8 * hl) * 512 + 16 * w + nr] = f2bf(qa[j]);
        __syncthreads();

        // scores[b][s] = scale * q[b] . k_proj[s,b]  (+ mask)
        {
            int b_ = tid >> 6, s = tid & 63;
            const float* kp = kproj + (size_t)(s * 16 + b_) * 512;
            float a = 0.0f;
            for (int k = 0; k < 512; ++k) a += bf2f(qbuf[b_ * 512 + k]) * kp[k];
            a *= 0.04419417382415922f;           // 1/sqrt(512)
            if (s >= src_len[b_]) a = -10000.0f; // MASK_FILL
            sc[b_ * 64 + s] = a;
        }
        __syncthreads();
        if (tid < 16) {
            float m = -3.4e38f;
            for (int s = 0; s < 64; ++s) m = fmaxf(m, sc[tid * 64 + s]);
            float sum = 0.0f;
            for (int s = 0; s < 64; ++s) { float e = __expf(sc[tid * 64 + s] - m); wts[tid * 64 + s] = e; sum += e; }
            float inv = 1.0f / sum;
            for (int s = 0; s < 64; ++s) wts[tid * 64 + s] *= inv;
        }
        __syncthreads();
        // ctx[b][h] = sum_s wts[b][s] * enc_out[s,b,h]  (used at step t+1)
        for (int i = tid; i < 8192; i += 1024) {
            int b_ = i >> 9, hh = i & 511;
            float a = 0.0f;
            for (int s = 0; s < 64; ++s)
                a += wts[b_ * 64 + s] * bf2f(encout[(size_t)(s * 16 + b_) * 512 + hh]);
            ctxs[b_ * 512 + hh] = f2bf(a);
        }
        __syncthreads();
    }
}

// ---------------------------------------------------------------------------
// Small utility kernels
// ---------------------------------------------------------------------------
__global__ void cvt_f32_bf16(u16* __restrict__ dst, const float* __restrict__ src, long n) {
    long i = (long)blockIdx.x * blockDim.x + threadIdx.x;
    if (i < n) dst[i] = f2bf(src[i]);
}
__global__ void vec_add_f32(float* __restrict__ d, const float* __restrict__ a,
                            const float* __restrict__ b, int n) {
    int i = blockIdx.x * blockDim.x + threadIdx.x;
    if (i < n) d[i] = a[i] + b[i];
}
__global__ void res_add_bf16(u16* __restrict__ dst, const u16* __restrict__ src, int n) {
    int i = blockIdx.x * blockDim.x + threadIdx.x;
    if (i < n) dst[i] = f2bf(bf2f(dst[i]) + bf2f(src[i]));
}
// dst row r = t*16+b (T-major), gather embedding rows of tok[b*T+t], f32 -> bf16
__global__ void gather_emb(u16* __restrict__ dst, const float* __restrict__ emb,
                           const int* __restrict__ tok, int T) {
    int n = T * 16 * 512;
    int i = blockIdx.x * blockDim.x + threadIdx.x;
    if (i >= n) return;
    int e = i & 511, r = i >> 9;
    int b = r & 15, t = r >> 4;
    int token = tok[b * T + t];
    dst[i] = f2bf(emb[(size_t)token * 512 + e]);
}
// Wcat[n][k] = k < Kih ? wih[n][k] : whh[n][k-Kih]   (f32 -> bf16)
__global__ void concat_w(u16* __restrict__ dst, const float* __restrict__ wih,
                         const float* __restrict__ whh, int Kih, int Khh) {
    int K = Kih + Khh;
    long n = 2048L * K;
    for (long i = (long)blockIdx.x * blockDim.x + threadIdx.x; i < n;
         i += (long)gridDim.x * blockDim.x) {
        int k = (int)(i % K);
        int row = (int)(i / K);
        float v = (k < Kih) ? wih[(size_t)row * Kih + k] : whh[(size_t)row * Khh + (k - Kih)];
        dst[i] = f2bf(v);
    }
}
__global__ void write_scalar(float* p, float v) {
    if (blockIdx.x == 0 && threadIdx.x == 0) p[0] = v;
}

// ---------------------------------------------------------------------------
// Host launch
// ---------------------------------------------------------------------------
struct ParamsHost {
    const float *emb, *wq, *wk, *projw, *projb;
    // lstm idx: 0=enc_l0_fwd, 1=enc_l0_bwd, 2..4=enc_stack, 5..8=dec_rnns
    const float *wih[9], *whh[9], *bih[9], *bhh[9];
    const int *src, *src_len, *tgt, *tgt_len;
};

extern "C" void kernel_launch(void* const* d_in, const int* in_sizes, int n_in,
                              void* d_out, int out_size, void* d_ws, size_t ws_size,
                              hipStream_t stream)
{
    (void)n_in; (void)ws_size;
    ParamsHost P;
    auto F = [&](int i) { return (const float*)d_in[i]; };
    auto I = [&](int i) { return (const int*)d_in[i]; };
    bool sorted_keys = (in_sizes[0] != 32000 * 512);  // jax pytree (sorted): [0]=attn_wk (262144)
    if (sorted_keys) {
        P.wk = F(0); P.wq = F(1);
        for (int k = 0; k < 4; ++k) { int b = 2 + 4 * k;               // dec: bhh,bih,whh,wih
            P.bhh[5 + k] = F(b); P.bih[5 + k] = F(b + 1); P.whh[5 + k] = F(b + 2); P.wih[5 + k] = F(b + 3); }
        P.emb = F(18);
        { int b = 19; P.bhh[1] = F(b); P.bih[1] = F(b + 1); P.whh[1] = F(b + 2); P.wih[1] = F(b + 3); } // l0_bwd
        { int b = 23; P.bhh[0] = F(b); P.bih[0] = F(b + 1); P.whh[0] = F(b + 2); P.wih[0] = F(b + 3); } // l0_fwd
        for (int j = 0; j < 3; ++j) { int b = 27 + 4 * j;
            P.bhh[2 + j] = F(b); P.bih[2 + j] = F(b + 1); P.whh[2 + j] = F(b + 2); P.wih[2 + j] = F(b + 3); }
        P.projb = F(39); P.projw = F(40);
        P.src = I(41); P.src_len = I(42); P.tgt = I(43); P.tgt_len = I(44);
    } else {                                                            // insertion order
        P.emb = F(0);
        { int b = 1; P.wih[0] = F(b); P.whh[0] = F(b + 1); P.bih[0] = F(b + 2); P.bhh[0] = F(b + 3); }
        { int b = 5; P.wih[1] = F(b); P.whh[1] = F(b + 1); P.bih[1] = F(b + 2); P.bhh[1] = F(b + 3); }
        for (int j = 0; j < 3; ++j) { int b = 9 + 4 * j;
            P.wih[2 + j] = F(b); P.whh[2 + j] = F(b + 1); P.bih[2 + j] = F(b + 2); P.bhh[2 + j] = F(b + 3); }
        for (int k = 0; k < 4; ++k) { int b = 21 + 4 * k;
            P.wih[5 + k] = F(b); P.whh[5 + k] = F(b + 1); P.bih[5 + k] = F(b + 2); P.bhh[5 + k] = F(b + 3); }
        P.wq = F(37); P.wk = F(38); P.projw = F(39); P.projb = F(40);
        P.src = I(41); P.src_len = I(42); P.tgt = I(43); P.tgt_len = I(44);
    }

    // ---- workspace bump allocator ----
    char* wp = (char*)d_ws;
    auto alloc = [&](size_t bytes) -> void* {
        void* r = wp; wp += (bytes + 255) & ~(size_t)255; return r;
    };
    u16* projw_bf = (u16*)alloc(32000UL * 512 * 2);
    u16* wq_bf    = (u16*)alloc(512UL * 512 * 2);
    u16* wk_bf    = (u16*)alloc(512UL * 512 * 2);
    int wihK[5] = {512, 512, 1024, 512, 512};
    u16* wih_bf[5]; for (int i = 0; i < 5; ++i) wih_bf[i] = (u16*)alloc(2048UL * wihK[i] * 2);
    u16* whh_bf[5]; for (int i = 0; i < 5; ++i) whh_bf[i] = (u16*)alloc(2048UL * 512 * 2);
    int wcatK[4] = {1536, 1024, 1024, 1024};
    u16* wcat[4]; for (int i = 0; i < 4; ++i) wcat[i] = (u16*)alloc(2048UL * wcatK[i] * 2);
    float* bs[9]; for (int i = 0; i < 9; ++i) bs[i] = (float*)alloc(2048 * 4);
    u16* emb_src = (u16*)alloc(1024UL * 512 * 2);
    u16* emb_tgt = (u16*)alloc(1024UL * 512 * 2);
    float* gx    = (float*)alloc(1024UL * 2048 * 4);
    u16* ys_cat  = (u16*)alloc(1024UL * 1024 * 2);
    u16* ys0     = (u16*)alloc(1024UL * 512 * 2);
    u16* ys1     = (u16*)alloc(1024UL * 512 * 2);
    u16* ys2     = (u16*)alloc(1024UL * 512 * 2);
    float* kproj = (float*)alloc(1024UL * 512 * 4);
    u16* hs_top  = (u16*)alloc(1024UL * 512 * 2);

    auto cvt = [&](u16* d, const float* s, long n) {
        cvt_f32_bf16<<<(int)((n + 255) / 256), 256, 0, stream>>>(d, s, n);
    };
    // ---- weight prep (one HBM pass; everything then L2-resident) ----
    cvt(projw_bf, P.projw, 32000L * 512);
    cvt(wq_bf, P.wq, 512L * 512);
    cvt(wk_bf, P.wk, 512L * 512);
    for (int i = 0; i < 5; ++i) { cvt(wih_bf[i], P.wih[i], 2048L * wihK[i]); cvt(whh_bf[i], P.whh[i], 2048L * 512); }
    for (int k = 0; k < 4; ++k)
        concat_w<<<256, 256, 0, stream>>>(wcat[k], P.wih[5 + k], P.whh[5 + k], wcatK[k] - 512, 512);
    for (int i = 0; i < 9; ++i)
        vec_add_f32<<<8, 256, 0, stream>>>(bs[i], P.bih[i], P.bhh[i], 2048);
    gather_emb<<<(1024 * 512 + 255) / 256, 256, 0, stream>>>(emb_src, P.emb, P.src, 64);
    gather_emb<<<(1024 * 512 + 255) / 256, 256, 0, stream>>>(emb_tgt, P.emb, P.tgt, 64);

    auto gemm = [&](const u16* A, int lda, const u16* W, int ldb, const float* bias,
                    float* C, int ldc, int M, int N, int K, int remap) {
        int tiles = (M / 16) * (N / 16);
        gemm_bf16<<<(tiles + 7) / 8, 256, 0, stream>>>(A, lda, W, ldb, bias, C, ldc, M, N, K, remap);
    };

    // ---- encoder ----
    gemm(emb_src, 512, wih_bf[0], 512, bs[0], gx, 2048, 1024, 2048, 512, 0);
    enc_scan<<<1, 1024, 0, stream>>>(gx, whh_bf[0], ys_cat, 1024, 0, 64, 0);
    gemm(emb_src, 512, wih_bf[1], 512, bs[1], gx, 2048, 1024, 2048, 512, 0);
    enc_scan<<<1, 1024, 0, stream>>>(gx, whh_bf[1], ys_cat, 1024, 512, 64, 1);
    gemm(ys_cat, 1024, wih_bf[2], 1024, bs[2], gx, 2048, 1024, 2048, 1024, 0);
    enc_scan<<<1, 1024, 0, stream>>>(gx, whh_bf[2], ys0, 512, 0, 64, 0);
    gemm(ys0, 512, wih_bf[3], 512, bs[3], gx, 2048, 1024, 2048, 512, 0);
    enc_scan<<<1, 1024, 0, stream>>>(gx, whh_bf[3], ys1, 512, 0, 64, 0);
    res_add_bf16<<<(524288 + 255) / 256, 256, 0, stream>>>(ys1, ys0, 524288);
    gemm(ys1, 512, wih_bf[4], 512, bs[4], gx, 2048, 1024, 2048, 512, 0);
    enc_scan<<<1, 1024, 0, stream>>>(gx, whh_bf[4], ys2, 512, 0, 64, 0);
    res_add_bf16<<<(524288 + 255) / 256, 256, 0, stream>>>(ys2, ys1, 524288);

    // ---- attention keys: k_proj = enc_out @ Wk^T (f32) ----
    gemm(ys2, 512, wk_bf, 512, nullptr, kproj, 512, 1024, 512, 512, 0);

    // ---- decoder (persistent) ----
    decoder_kernel<<<1, 1024, 0, stream>>>(emb_tgt, wcat[0], wcat[1], wcat[2], wcat[3],
                                           bs[5], bs[6], bs[7], bs[8],
                                           wq_bf, kproj, ys2, P.src_len, hs_top);

    // ---- logits: hs_top(1024x512) @ proj_w^T (32000x512) + proj_b, (t,b)->(b,t) remap ----
    gemm(hs_top, 512, projw_bf, 512, P.projb, (float*)d_out, 32000, 1024, 32000, 512, 1);

    // reference returns (logits, Tt); if harness concatenates the scalar, write it
    if (out_size > 1024 * 32000)
        write_scalar<<<1, 1, 0, stream>>>((float*)d_out + 1024 * 32000, 64.0f);
}
